// GT_7327214207519
// MI455X (gfx1250) — matrix operand order repaired and checked by reading, not compiled, not run on hardware
//
#include <hip/hip_runtime.h>
#include <hip/hip_bf16.h>

// ---------------------------------------------------------------------------
// Graph Transformer (2x TransformerConv, H=4, C=128) for MI455X / gfx1250.
// Dense projections: bf16 WMMA (v_wmma_f32_16x16x32_bf16), f32 accumulate,
// 16x64 register-blocked tiles, compile-time K, pipelined b128 loads.
// Edge attention: CSR-by-destination (built once, shared by both layers),
// node-centric wave32 kernel -> zero atomics in the hot path, plain stores.
// ---------------------------------------------------------------------------

#define NNODES 20000
#define NEDGES 320000
#define DIN    128
#define HID    128
#define OUTC   128
#define NHEAD  4
#define HHID   (NHEAD * HID)     // 512
#define SCALE_QK 0.08838834764831845f  // 1/sqrt(128)

typedef __attribute__((ext_vector_type(16))) __bf16 v16bf;
typedef __attribute__((ext_vector_type(8)))  float  v8f;

// ---------------------------------------------------------------- utilities
__global__ void k_fill_u32(unsigned* p, unsigned val, int n) {
    int t = blockIdx.x * blockDim.x + threadIdx.x;
    if (t < n) p[t] = val;
}

__global__ void k_cvt_f32_bf16(const float* __restrict__ src, __bf16* __restrict__ dst, int n) {
    int t = blockIdx.x * blockDim.x + threadIdx.x;
    if (t < n) dst[t] = (__bf16)src[t];
}

// ------------------------------------------------------------- CSR building
__global__ void k_hist(const int* __restrict__ dst, unsigned* __restrict__ count) {
    int e = blockIdx.x * blockDim.x + threadIdx.x;
    if (e < NEDGES) atomicAdd(&count[dst[e]], 1u);
}

// Single-block exclusive scan over NNODES counts (LDS Hillis-Steele, chunked).
__global__ void k_scan(const unsigned* __restrict__ count, unsigned* __restrict__ rowptr) {
    __shared__ unsigned sh[1024];
    __shared__ unsigned s_carry;
    const int tid = threadIdx.x;
    if (tid == 0) s_carry = 0;
    __syncthreads();
    for (int base = 0; base < NNODES; base += 1024) {
        const unsigned carry = s_carry;
        const int i = base + tid;
        const unsigned v = (i < NNODES) ? count[i] : 0u;
        sh[tid] = v;
        __syncthreads();
        for (int ofs = 1; ofs < 1024; ofs <<= 1) {
            unsigned t = (tid >= ofs) ? sh[tid - ofs] : 0u;
            __syncthreads();
            sh[tid] += t;
            __syncthreads();
        }
        if (i < NNODES) rowptr[i] = carry + sh[tid] - v;   // exclusive
        const unsigned tot = sh[1023];
        __syncthreads();
        if (tid == 0) s_carry = carry + tot;
        __syncthreads();
    }
    if (tid == 0) rowptr[NNODES] = s_carry;
}

__global__ void k_copy_u32(const unsigned* __restrict__ src, unsigned* __restrict__ dst, int n) {
    int t = blockIdx.x * blockDim.x + threadIdx.x;
    if (t < n) dst[t] = src[t];
}

__global__ void k_build_csr(const int* __restrict__ src, const int* __restrict__ dst,
                            unsigned* __restrict__ cursor, unsigned* __restrict__ ecol) {
    int e = blockIdx.x * blockDim.x + threadIdx.x;
    if (e >= NEDGES) return;
    unsigned pos = atomicAdd(&cursor[dst[e]], 1u);
    ecol[pos] = (unsigned)src[e];
}

// -------------------------------------------------- weight repack (one-shot)
// Pack row-major f32 W[K x Ncols] into bf16 WMMA B-fragment-major layout:
// frag index f = (nt*ksteps + ks)*32 + lane ; 16 contiguous bf16 per lane.
// B-matrix 32x16 layout (wave32): lane 0-15 -> n, K=0..7,16..23 ; lane 16-31 -> K=8..15,24..31.
__global__ void k_repack_w(const float* __restrict__ W, __bf16* __restrict__ out,
                           int K, int Ncols) {
    int ksteps = K >> 5, ntiles = Ncols >> 4;
    int t = blockIdx.x * blockDim.x + threadIdx.x;
    int total = ntiles * ksteps * 32;
    if (t >= total) return;
    int lane = t & 31;
    int ks   = (t >> 5) % ksteps;
    int nt   = (t >> 5) / ksteps;
    int n  = nt * 16 + (lane & 15);
    int kb = ks * 32 + ((lane >> 4) << 3);
    __bf16* o = out + (size_t)t * 16;
#pragma unroll
    for (int v = 0; v < 8; ++v) {
        o[2*v]     = (__bf16)W[(size_t)(kb + 2*v + ((v >> 2) << 3))     * Ncols + n];
        o[2*v + 1] = (__bf16)W[(size_t)(kb + 2*v + ((v >> 2) << 3) + 1) * Ncols + n];
    }
}

// ---------------------------------------------------------------- WMMA GEMM
// C[M x Ncols] = A[M x K](bf16, row-major) @ Bpack + bias.
// One wave computes a 16x64 tile (4 accumulators). Compile-time K, fully
// unrolled k-loop, double-buffered fragment loads.
// Requires M%16==0, Ncols%64==0, K in {128, 512}.
template <int OUT_BF16, int KK>
__global__ void k_gemm_wmma(const __bf16* __restrict__ A,
                            const __bf16* __restrict__ Bp,
                            const float*  __restrict__ bias,
                            void* __restrict__ Cout,
                            int M, int Ncols) {
    constexpr int ksteps  = KK >> 5;
    constexpr int bstride = ksteps * 512;          // elements between column tiles
    const int lane    = threadIdx.x & 31;
    const int wid     = (blockIdx.x * blockDim.x + threadIdx.x) >> 5;
    const int ngroups = Ncols >> 6;                // 64-wide column groups
    const int mtiles  = M >> 4;
    if (wid >= mtiles * ngroups) return;           // wave-uniform: EXEC stays all-1s
    const int mt = wid / ngroups;
    const int ng = wid % ngroups;

    // A fragment (16-bit A 16x32): lane m = lane&15 ; K-half = (lane>>4)*8
    const int m  = lane & 15;
    const int kh = (lane >> 4) << 3;
    const __bf16* ap = A + (size_t)(mt * 16 + m) * KK + kh;
    const __bf16* bp = Bp + ((size_t)(ng * 4) * ksteps * 32 + lane) * 16;

    union AB { v16bf v; uint4 q[2]; };
    v8f acc0 = {}, acc1 = {}, acc2 = {}, acc3 = {};

    // prologue: k-step 0 fragments
    AB a;
    a.q[0] = *(const uint4*)(ap);
    a.q[1] = *(const uint4*)(ap + 16);
    v16bf b0 = *(const v16bf*)(bp);
    v16bf b1 = *(const v16bf*)(bp + bstride);
    v16bf b2 = *(const v16bf*)(bp + 2 * bstride);
    v16bf b3 = *(const v16bf*)(bp + 3 * bstride);

#pragma unroll
    for (int ks = 0; ks < ksteps - 1; ++ks) {
        // issue next k-step's loads before consuming current fragments
        AB an;
        an.q[0] = *(const uint4*)(ap + 32);
        an.q[1] = *(const uint4*)(ap + 48);
        v16bf n0 = *(const v16bf*)(bp + 512);
        v16bf n1 = *(const v16bf*)(bp + 512 + bstride);
        v16bf n2 = *(const v16bf*)(bp + 512 + 2 * bstride);
        v16bf n3 = *(const v16bf*)(bp + 512 + 3 * bstride);
        __builtin_prefetch(ap + 64, 0, 1);         // global_prefetch_b8 on streamed A

        acc0 = __builtin_amdgcn_wmma_f32_16x16x32_bf16(false, a.v, false, b0, (short)0, acc0, false, false);
        acc1 = __builtin_amdgcn_wmma_f32_16x16x32_bf16(false, a.v, false, b1, (short)0, acc1, false, false);
        acc2 = __builtin_amdgcn_wmma_f32_16x16x32_bf16(false, a.v, false, b2, (short)0, acc2, false, false);
        acc3 = __builtin_amdgcn_wmma_f32_16x16x32_bf16(false, a.v, false, b3, (short)0, acc3, false, false);

        a = an; b0 = n0; b1 = n1; b2 = n2; b3 = n3;
        ap += 32;
        bp += 512;
    }
    // epilogue: last k-step
    acc0 = __builtin_amdgcn_wmma_f32_16x16x32_bf16(false, a.v, false, b0, (short)0, acc0, false, false);
    acc1 = __builtin_amdgcn_wmma_f32_16x16x32_bf16(false, a.v, false, b1, (short)0, acc1, false, false);
    acc2 = __builtin_amdgcn_wmma_f32_16x16x32_bf16(false, a.v, false, b2, (short)0, acc2, false, false);
    acc3 = __builtin_amdgcn_wmma_f32_16x16x32_bf16(false, a.v, false, b3, (short)0, acc3, false, false);

    const int rbase = mt * 16 + ((lane >> 4) << 3);
    const int nlo   = lane & 15;
    v8f accs[4] = {acc0, acc1, acc2, acc3};
#pragma unroll
    for (int j = 0; j < 4; ++j) {
        const int col = (ng * 4 + j) * 16 + nlo;
        const float bv = bias[col];
        if (OUT_BF16) {
            __bf16* C = (__bf16*)Cout;
#pragma unroll
            for (int v = 0; v < 8; ++v)
                C[(size_t)(rbase + v) * Ncols + col] = (__bf16)(accs[j][v] + bv);
        } else {
            float* C = (float*)Cout;
#pragma unroll
            for (int v = 0; v < 8; ++v)
                C[(size_t)(rbase + v) * Ncols + col] = accs[j][v] + bv;
        }
    }
}

// -------------------------------------------- node-centric edge attention
// One wave per (node, head). q[node] register-resident; three short passes
// over the node's CSR edge list: (1) logits + running max (channel-parallel
// dot + wave butterfly), (2) edge-parallel exp + sum, (3) channel-parallel
// weighted gather of v[src]. No atomics; one plain 16B store per lane.
__global__ void k_attn_node(const __bf16* __restrict__ qb,
                            const __bf16* __restrict__ kb,
                            const __bf16* __restrict__ vb,
                            const unsigned* __restrict__ rowptr,
                            const unsigned* __restrict__ ecol,
                            float* __restrict__ alpha,
                            float* __restrict__ attn) {
    const int lane = threadIdx.x & 31;
    const int wid  = (blockIdx.x * blockDim.x + threadIdx.x) >> 5;
    if (wid >= NNODES * NHEAD) return;
    const int d = wid >> 2, h = wid & 3;
    const int r0 = (int)rowptr[d], r1 = (int)rowptr[d + 1];

    union B4 { uint2 u; __bf16 b[4]; };
    B4 qv;
    qv.u = *(const uint2*)(qb + (size_t)d * HHID + h * HID + lane * 4);
    const float q0 = (float)qv.b[0], q1 = (float)qv.b[1];
    const float q2 = (float)qv.b[2], q3 = (float)qv.b[3];

    // pass 1: logits + max
    float mx = -3.0e38f;
    for (int p = r0; p < r1; ++p) {
        const int s = (int)ecol[p];
        B4 kv;
        kv.u = *(const uint2*)(kb + (size_t)s * HHID + h * HID + lane * 4);
        float acc = q0 * (float)kv.b[0] + q1 * (float)kv.b[1]
                  + q2 * (float)kv.b[2] + q3 * (float)kv.b[3];
#pragma unroll
        for (int off = 16; off; off >>= 1) acc += __shfl_xor(acc, off, 32);
        const float l = acc * SCALE_QK;              // all lanes hold l
        if (lane == 0) alpha[(size_t)p * NHEAD + h] = l;
        mx = fmaxf(mx, l);
    }

    // pass 2: edge-parallel exp + segment sum
    float part = 0.f;
    for (int p = r0 + lane; p < r1; p += 32) {
        float a = __expf(alpha[(size_t)p * NHEAD + h] - mx);
        alpha[(size_t)p * NHEAD + h] = a;
        part += a;
    }
#pragma unroll
    for (int off = 16; off; off >>= 1) part += __shfl_xor(part, off, 32);
    const float inv = 1.f / (part + 1e-16f);

    // pass 3: channel-parallel weighted accumulate of v[src]
    float a0 = 0.f, a1 = 0.f, a2 = 0.f, a3 = 0.f;
    for (int p = r0; p < r1; ++p) {
        const int s = (int)ecol[p];
        const float w = alpha[(size_t)p * NHEAD + h] * inv;
        B4 vv;
        vv.u = *(const uint2*)(vb + (size_t)s * HHID + h * HID + lane * 4);
        a0 += w * (float)vv.b[0];
        a1 += w * (float)vv.b[1];
        a2 += w * (float)vv.b[2];
        a3 += w * (float)vv.b[3];
    }
    float4 o = make_float4(a0, a1, a2, a3);
    *(float4*)(attn + (size_t)d * HHID + h * HID + lane * 4) = o;
}

// ------------------------------- layer-0 combine: h = relu(attn+skip) -> bf16
__global__ void k_relu_add_bf16(const float* __restrict__ attn,
                                const float* __restrict__ skip,
                                __bf16* __restrict__ hb, int n) {
    int t = blockIdx.x * blockDim.x + threadIdx.x;
    if (t >= n) return;
    float v = attn[t] + skip[t];
    hb[t] = (__bf16)(v > 0.f ? v : 0.f);
}

// -------------------------- layer-1 combine: out += mean over heads of attn
__global__ void k_final_mean(const float* __restrict__ attn,
                             float* __restrict__ out, int n) {
    int t = blockIdx.x * blockDim.x + threadIdx.x;
    if (t >= n) return;
    int node = t >> 7, c = t & 127;
    const float* a = attn + (size_t)node * HHID + c;
    out[t] += 0.25f * (a[0] + a[128] + a[256] + a[384]);
}

// ---------------------------------------------------------------------------
static inline int cdiv(long long a, long long b) { return (int)((a + b - 1) / b); }

extern "C" void kernel_launch(void* const* d_in, const int* in_sizes, int n_in,
                              void* d_out, int out_size, void* d_ws, size_t ws_size,
                              hipStream_t stream) {
    const float* x   = (const float*)d_in[0];
    const int*   ei  = (const int*)d_in[1];
    const float* Wq0 = (const float*)d_in[2];  const float* bq0 = (const float*)d_in[3];
    const float* Wk0 = (const float*)d_in[4];  const float* bk0 = (const float*)d_in[5];
    const float* Wv0 = (const float*)d_in[6];  const float* bv0 = (const float*)d_in[7];
    const float* Ws0 = (const float*)d_in[8];  const float* bs0 = (const float*)d_in[9];
    const float* Wq1 = (const float*)d_in[10]; const float* bq1 = (const float*)d_in[11];
    const float* Wk1 = (const float*)d_in[12]; const float* bk1 = (const float*)d_in[13];
    const float* Wv1 = (const float*)d_in[14]; const float* bv1 = (const float*)d_in[15];
    const float* Ws1 = (const float*)d_in[16]; const float* bs1 = (const float*)d_in[17];
    float* out = (float*)d_out;
    const int* src = ei;
    const int* dst = ei + NEDGES;

    // ---- workspace carve-up (256B aligned) --------------------------------
    char* base = (char*)d_ws;
    size_t off = 0;
    auto alloc = [&](size_t bytes) -> void* {
        void* p = base + off;
        off += (bytes + 255) & ~(size_t)255;
        return p;
    };
    __bf16* xb   = (__bf16*)alloc((size_t)NNODES * DIN  * 2);
    __bf16* hb   = (__bf16*)alloc((size_t)NNODES * HHID * 2);
    __bf16* qb   = (__bf16*)alloc((size_t)NNODES * HHID * 2);
    __bf16* kb   = (__bf16*)alloc((size_t)NNODES * HHID * 2);
    __bf16* vb   = (__bf16*)alloc((size_t)NNODES * HHID * 2);
    float*  skip0= (float*) alloc((size_t)NNODES * HHID * 4);
    float*  attn = (float*) alloc((size_t)NNODES * HHID * 4);
    float*  alpha= (float*) alloc((size_t)NEDGES * NHEAD * 4);
    unsigned* count  = (unsigned*)alloc((size_t)NNODES * 4);
    unsigned* rowptr = (unsigned*)alloc((size_t)(NNODES + 1) * 4);
    unsigned* cursor = (unsigned*)alloc((size_t)NNODES * 4);
    unsigned* ecol   = (unsigned*)alloc((size_t)NEDGES * 4);
    __bf16* pq0 = (__bf16*)alloc((size_t)DIN  * HHID * 2);
    __bf16* pk0 = (__bf16*)alloc((size_t)DIN  * HHID * 2);
    __bf16* pv0 = (__bf16*)alloc((size_t)DIN  * HHID * 2);
    __bf16* ps0 = (__bf16*)alloc((size_t)DIN  * HHID * 2);
    __bf16* pq1 = (__bf16*)alloc((size_t)HHID * HHID * 2);
    __bf16* pk1 = (__bf16*)alloc((size_t)HHID * HHID * 2);
    __bf16* pv1 = (__bf16*)alloc((size_t)HHID * HHID * 2);
    __bf16* ps1 = (__bf16*)alloc((size_t)HHID * OUTC * 2);

    // ---- one-shot conversions / repacks / CSR -----------------------------
    {
        int n = NNODES * DIN;
        k_cvt_f32_bf16<<<cdiv(n, 256), 256, 0, stream>>>(x, xb, n);
    }
    auto repack = [&](const float* W, __bf16* P, int K, int Nc) {
        int total = (Nc >> 4) * (K >> 5) * 32;
        k_repack_w<<<cdiv(total, 256), 256, 0, stream>>>(W, P, K, Nc);
    };
    repack(Wq0, pq0, DIN, HHID);  repack(Wk0, pk0, DIN, HHID);
    repack(Wv0, pv0, DIN, HHID);  repack(Ws0, ps0, DIN, HHID);
    repack(Wq1, pq1, HHID, HHID); repack(Wk1, pk1, HHID, HHID);
    repack(Wv1, pv1, HHID, HHID); repack(Ws1, ps1, HHID, OUTC);

    // CSR by destination: histogram -> exclusive scan -> stable-ish scatter
    k_fill_u32 <<<cdiv(NNODES, 256), 256, 0, stream>>>(count, 0u, NNODES);
    k_hist     <<<cdiv(NEDGES, 256), 256, 0, stream>>>(dst, count);
    k_scan     <<<1, 1024, 0, stream>>>(count, rowptr);
    k_copy_u32 <<<cdiv(NNODES, 256), 256, 0, stream>>>(rowptr, cursor, NNODES);
    k_build_csr<<<cdiv(NEDGES, 256), 256, 0, stream>>>(src, dst, cursor, ecol);

    auto gemm128_bf16 = [&](const __bf16* A, const __bf16* P, const float* bias,
                            __bf16* C, int M, int Nc) {
        int waves = (M >> 4) * (Nc >> 6);
        k_gemm_wmma<1, 128><<<cdiv((long long)waves * 32, 256), 256, 0, stream>>>(A, P, bias, C, M, Nc);
    };
    auto gemm128_f32 = [&](const __bf16* A, const __bf16* P, const float* bias,
                           float* C, int M, int Nc) {
        int waves = (M >> 4) * (Nc >> 6);
        k_gemm_wmma<0, 128><<<cdiv((long long)waves * 32, 256), 256, 0, stream>>>(A, P, bias, C, M, Nc);
    };
    auto gemm512_bf16 = [&](const __bf16* A, const __bf16* P, const float* bias,
                            __bf16* C, int M, int Nc) {
        int waves = (M >> 4) * (Nc >> 6);
        k_gemm_wmma<1, 512><<<cdiv((long long)waves * 32, 256), 256, 0, stream>>>(A, P, bias, C, M, Nc);
    };
    auto gemm512_f32 = [&](const __bf16* A, const __bf16* P, const float* bias,
                           float* C, int M, int Nc) {
        int waves = (M >> 4) * (Nc >> 6);
        k_gemm_wmma<0, 512><<<cdiv((long long)waves * 32, 256), 256, 0, stream>>>(A, P, bias, C, M, Nc);
    };
    auto attn_layer = [&](const __bf16* q, const __bf16* k, const __bf16* v, float* at) {
        long long waves = (long long)NNODES * NHEAD;
        k_attn_node<<<cdiv(waves * 32, 256), 256, 0, stream>>>(q, k, v, rowptr, ecol, alpha, at);
    };

    // ---- layer 0: TransformerConv(128 -> 128, H=4, concat) ----------------
    gemm128_bf16(xb, pq0, bq0, qb, NNODES, HHID);
    gemm128_bf16(xb, pk0, bk0, kb, NNODES, HHID);
    gemm128_bf16(xb, pv0, bv0, vb, NNODES, HHID);
    gemm128_f32 (xb, ps0, bs0, skip0, NNODES, HHID);
    attn_layer(qb, kb, vb, attn);
    {
        int n = NNODES * HHID;
        k_relu_add_bf16<<<cdiv(n, 256), 256, 0, stream>>>(attn, skip0, hb, n);
    }

    // ---- layer 1: TransformerConv(512 -> 128, H=4, mean) ------------------
    gemm512_bf16(hb, pq1, bq1, qb, NNODES, HHID);
    gemm512_bf16(hb, pk1, bk1, kb, NNODES, HHID);
    gemm512_bf16(hb, pv1, bv1, vb, NNODES, HHID);
    gemm512_f32 (hb, ps1, bs1, out, NNODES, OUTC);   // skip path straight to d_out
    attn_layer(qb, kb, vb, attn);
    {
        int n = NNODES * OUTC;
        k_final_mean<<<cdiv(n, 256), 256, 0, stream>>>(attn, out, n);
    }
}
